// MPNetForPretraining_56745107915290
// MI455X (gfx1250) — matrix operand order repaired and checked by reading, not compiled, not run on hardware
//
#include <hip/hip_runtime.h>
#include <hip/hip_bf16.h>
#include <stdint.h>

// ---------------- CDNA5 / gfx1250 types ----------------
typedef __bf16 bf16_t;
typedef __attribute__((ext_vector_type(16))) __bf16 v16bf;
typedef __attribute__((ext_vector_type(8)))  __bf16 v8bf;
typedef __attribute__((ext_vector_type(8)))  float  v8f;
typedef __attribute__((ext_vector_type(4)))  int    v4i;

union ABfrag { v16bf v; v8bf h[2]; };

#if defined(__AMDGCN__) && __has_builtin(__builtin_amdgcn_global_load_async_to_lds_b128) && __has_builtin(__builtin_amdgcn_s_wait_asynccnt)
#define USE_ASYNC_LDS 1
typedef __attribute__((address_space(1))) v4i* gv4i_p;   // global int4*
typedef __attribute__((address_space(3))) v4i* lv4i_p;   // LDS int4*
#else
#define USE_ASYNC_LDS 0
#endif

// ---------------- model constants ----------------
#define BB   4
#define SS   512
#define PP   80
#define EE   768
#define HH   12
#define DHH  64
#define FFNN 3072
#define NL   12
#define VV   30527
#define LC   592          // S+P   (content stream length / key length)
#define LT   672          // S+2P  (content + query rows)
#define MT   (BB*LT)      // 2688 total token rows
#define SCALE_Q 0.125f    // 64^-0.5

__device__ __forceinline__ bf16_t to_bf16(float f) {
  unsigned u = __builtin_bit_cast(unsigned, f);
  unsigned r = u + 0x7FFFu + ((u >> 16) & 1u);      // round-to-nearest-even
  unsigned short h = (unsigned short)(r >> 16);
  return __builtin_bit_cast(bf16_t, h);
}

__device__ __forceinline__ v8f zero8() {
  v8f z = {0.f,0.f,0.f,0.f,0.f,0.f,0.f,0.f};
  return z;
}

__device__ __forceinline__ v8f wmma_bf16(v16bf a, v16bf b, v8f c) {
  return __builtin_amdgcn_wmma_f32_16x16x32_bf16(false, a, false, b, (short)0, c, false, false);
}

__device__ __forceinline__ float gelu_exact(float x) {
  return 0.5f * x * (1.0f + erff(x * 0.70710678118654752f));
}

// ---------------- fp32 -> bf16 conversion ----------------
__global__ __launch_bounds__(256) void cvt_bf16_kernel(const float* __restrict__ src,
                                                       bf16_t* __restrict__ dst, long n) {
  long i = (long)blockIdx.x * 256 + threadIdx.x;
  long stride = (long)gridDim.x * 256;
  for (; i < n; i += stride) dst[i] = to_bf16(src[i]);
}

// ---------------- embedding: tok_emb[id] + pos_emb[pos] ----------------
__global__ __launch_bounds__(256) void embed_kernel(const int* __restrict__ ids,
                                                    const float* __restrict__ tok,
                                                    const float* __restrict__ pos,
                                                    float* __restrict__ X) {
  int row = blockIdx.x;                 // 0..MT-1
  int b = row / LT, t = row % LT;
  int p = (t < SS) ? t : ((t < LC) ? t - PP : t - 2*PP);
  int id = ids[b * LT + t];
  const float* te = tok + (size_t)id * EE;
  const float* pe = pos + (size_t)p * EE;
  float* xo = X + (size_t)row * EE;
  for (int c = threadIdx.x; c < EE; c += blockDim.x) xo[c] = te[c] + pe[c];
}

// ---------------- layernorm over E=768 (one row per block) ----------------
__global__ __launch_bounds__(256) void ln_kernel(const float* __restrict__ in,
                                                 const float* __restrict__ g,
                                                 const float* __restrict__ bsh,
                                                 float* __restrict__ outF,
                                                 bf16_t* __restrict__ outB) {
  __shared__ float redS[8], redQ[8];
  const int row = blockIdx.x;
  const int tid = threadIdx.x, lane = tid & 31, wave = tid >> 5;
  const float* x = in + (size_t)row * EE;
  float v0 = x[tid], v1 = x[tid + 256], v2 = x[tid + 512];
  float s1 = v0 + v1 + v2;
  float s2 = v0*v0 + v1*v1 + v2*v2;
  for (int o = 16; o; o >>= 1) { s1 += __shfl_xor(s1, o, 32); s2 += __shfl_xor(s2, o, 32); }
  if (lane == 0) { redS[wave] = s1; redQ[wave] = s2; }
  __syncthreads();
  if (tid == 0) {
    float a = 0.f, b2 = 0.f;
    for (int i = 0; i < 8; i++) { a += redS[i]; b2 += redQ[i]; }
    redS[0] = a; redQ[0] = b2;
  }
  __syncthreads();
  float mean = redS[0] * (1.0f / EE);
  float var  = redQ[0] * (1.0f / EE) - mean * mean;
  float inv  = rsqrtf(var + 1e-5f);
  for (int k = 0; k < 3; k++) {
    int c = tid + k * 256;
    float v = (k == 0 ? v0 : (k == 1 ? v1 : v2));
    float y = (v - mean) * inv * g[c] + bsh[c];
    if (outF) outF[(size_t)row * EE + c] = y;
    if (outB) outB[(size_t)row * EE + c] = to_bf16(y);
  }
}

// ---------------- additive attention bias (rel-pos bucket + masks) ----------------
__device__ __forceinline__ bool attn_masked(int i, int k) {
  if (i < LC) {                              // content-stream query
    if (k < SS - PP) return false;
    if (k < SS)      return !(i >= SS - PP && i < SS && k <= i);
    return (i >= SS - PP && i < SS && (k - PP) <= i);
  } else {                                   // query-stream query
    int iq = i - LC;
    if (k < SS - PP) return false;
    if (k < SS)      return !((k - (SS - PP)) < iq);
    return !((k - SS) >= iq);
  }
}

__global__ __launch_bounds__(256) void bias_kernel(const float* __restrict__ rel_bias,
                                                   float* __restrict__ biasM) {
  int idx = blockIdx.x;            // h*LT + i
  int i = idx % LT, h = idx / LT;
  int qp = (i < SS) ? i : ((i < LC) ? i - PP : i - 2*PP);
  for (int k = threadIdx.x; k < LC; k += blockDim.x) {
    int kp = (k < SS) ? k : k - PP;
    int rel = kp - qp;
    int ret = (rel > 0) ? 16 : 0;
    int n = (rel < 0) ? -rel : rel;
    int ib;
    if (n < 8) ib = n;
    else {
      int lg = 8 + (int)(logf((float)n * 0.125f) * (8.0f / logf(16.0f)));
      ib = lg > 15 ? 15 : lg;
    }
    float v = attn_masked(i, k) ? -3.0e38f : rel_bias[(ret + ib) * HH + h];
    biasM[((size_t)h * LT + i) * LC + k] = v;
  }
}

// ---------------- WMMA GEMM:  out = act((A @ W^T + bias) * scale) (+residual) ----------------
// A: bf16 [M,K] lda ; W: bf16 [N,K] ldw
// block tile 64x256, wave tile 32x64 (acc 2x4), K step 32, double-buffered async A staging
__global__ __launch_bounds__(256) void gemm_kernel(
    const bf16_t* __restrict__ A, int lda,
    const bf16_t* __restrict__ W, int ldw,
    const float* __restrict__ bias,
    const float* __restrict__ residual,
    float* __restrict__ outF, int ldo,
    bf16_t* __restrict__ outB, int bmode,     // 0 none, 1 row-major, 2 per-batch transposed (672 inner)
    int M, int N, int K, float scale, int act)
{
  __shared__ bf16_t Atile[2][64][32];
  const int tid  = threadIdx.x;
  const int lane = tid & 31, wave = tid >> 5;
  const int wr = wave >> 2, wc = wave & 3;
  const int rowBlock = blockIdx.y * 64;
  const int colBlock = blockIdx.x * 256;
  const int r = lane & 15, g = lane >> 4;

  v8f acc[2][4];
  for (int m = 0; m < 2; m++)
    for (int n = 0; n < 4; n++) acc[m][n] = zero8();

  const bf16_t* wp[4];
  for (int n = 0; n < 4; n++) {
    int col = colBlock + wc * 64 + n * 16 + r;
    if (col >= N) col = N - 1;                       // clamp loads; stores are guarded
    wp[n] = W + (size_t)col * ldw + g * 8;
  }
  const int arow = tid >> 2, ac = (tid & 3) * 8;     // 256 threads stage 64x32 bf16
  const bf16_t* aptr = A + (size_t)(rowBlock + arow) * lda + ac;

  const int nk = K >> 5;

#if USE_ASYNC_LDS
  // prologue: kick off first A tile
  __builtin_amdgcn_global_load_async_to_lds_b128((gv4i_p)aptr,
                                                 (lv4i_p)&Atile[0][arow][ac], 0, 0);
#endif
  for (int ks = 0; ks < nk; ks++) {
    const int k = ks * 32;
#if USE_ASYNC_LDS
    const int cur = ks & 1;
    __builtin_amdgcn_s_wait_asynccnt(0);             // our chunk of buffer `cur` arrived
    __syncthreads();                                 // everyone's chunk arrived; prev reads of cur^1 done
    // unconditional prefetch of the next tile (last iter re-fetches a valid tile into the
    // dead buffer; never read, drained by endpgm's implicit wait-idle) -> no branch in loop
    {
      const int kn = (ks + 1 < nk) ? (k + 32) : k;
      __builtin_amdgcn_global_load_async_to_lds_b128((gv4i_p)(aptr + kn),
                                                     (lv4i_p)&Atile[cur ^ 1][arow][ac], 0, 0);
    }
#else
    const int cur = 0;
    __syncthreads();
    *(v8bf*)&Atile[0][arow][ac] = *(const v8bf*)(aptr + k);
    __syncthreads();
#endif

    // pre-load all B fragments so the four b128 pairs pipeline
    ABfrag bfr[4];
    for (int n = 0; n < 4; n++) {
      const bf16_t* p = wp[n] + k;
      bfr[n].h[0] = *(const v8bf*)p;
      bfr[n].h[1] = *(const v8bf*)(p + 16);
    }
    ABfrag a[2];
    for (int m = 0; m < 2; m++) {
      int lr = wr * 32 + m * 16 + r;
      a[m].h[0] = *(const v8bf*)&Atile[cur][lr][g * 8];
      a[m].h[1] = *(const v8bf*)&Atile[cur][lr][16 + g * 8];
    }
    for (int n = 0; n < 4; n++) {
      acc[0][n] = wmma_bf16(a[0].v, bfr[n].v, acc[0][n]);
      acc[1][n] = wmma_bf16(a[1].v, bfr[n].v, acc[1][n]);
    }
  }

  for (int n = 0; n < 4; n++) {
    int col = colBlock + wc * 64 + n * 16 + r;
    if (col >= N) continue;
    float bv = bias ? bias[col] : 0.0f;
    for (int m = 0; m < 2; m++) {
      for (int j = 0; j < 8; j++) {
        int row = rowBlock + wr * 32 + m * 16 + ((lane < 16) ? j : 8 + j);
        float v = (acc[m][n][j] + bv) * scale;
        if (act == 1) v = gelu_exact(v);
        if (residual) v += residual[(size_t)row * ldo + col];
        if (outF) outF[(size_t)row * ldo + col] = v;
        if (outB) {
          if (bmode == 1) outB[(size_t)row * ldo + col] = to_bf16(v);
          else {
            int batch = row / LT, ir = row % LT;     // V transposed: [b][e][672]
            outB[((size_t)batch * EE + col) * LT + ir] = to_bf16(v);
          }
        }
      }
    }
  }
}

// ---------------- fused attention: scores(WMMA) + bias/mask + softmax + P@V(WMMA) ----------------
// grid: (42 q-tiles, H, B); Q,K row-major bf16 [B*LT,E]; Vt bf16 [B][E][LT]; Ob bf16 [B*LT,E]
__global__ __launch_bounds__(256) void attn_kernel(const bf16_t* __restrict__ Qb,
                                                   const bf16_t* __restrict__ Kb,
                                                   const bf16_t* __restrict__ Vt,
                                                   const float* __restrict__ biasM,
                                                   bf16_t* __restrict__ Ob)
{
  __shared__ float  Sc[16][608];
  __shared__ bf16_t Pm[16][608];
  const int b = blockIdx.z, h = blockIdx.y, qt = blockIdx.x;
  const int tid = threadIdx.x, lane = tid & 31, wave = tid >> 5;
  const int r = lane & 15, g = lane >> 4;
  const int qrow0 = qt * 16;

  { // pad keys 592..607 so 19*32 K-steps work out
    int pr = tid >> 4, pc = 592 + (tid & 15);
    Sc[pr][pc] = -3.0e38f;
  }

  // Q fragments (already scaled by SCALE_Q in the Q projection)
  ABfrag aQ[2];
  const bf16_t* qp = Qb + ((size_t)(b * LT + qrow0 + r)) * EE + h * DHH;
  for (int s = 0; s < 2; s++) {
    aQ[s].h[0] = *(const v8bf*)(qp + s * 32 + g * 8);
    aQ[s].h[1] = *(const v8bf*)(qp + s * 32 + 16 + g * 8);
  }

  // scores: 37 key tiles distributed over 8 waves
  for (int kt = wave; kt < 37; kt += 8) {
    v8f acc = zero8();
    int key = kt * 16 + r;
    const bf16_t* kp = Kb + ((size_t)(b * LT + key)) * EE + h * DHH + g * 8;
    for (int s = 0; s < 2; s++) {
      ABfrag bf;
      bf.h[0] = *(const v8bf*)(kp + s * 32);
      bf.h[1] = *(const v8bf*)(kp + s * 32 + 16);
      acc = wmma_bf16(aQ[s].v, bf.v, acc);
    }
    const float* bp = biasM + ((size_t)h * LT + qrow0) * LC + key;
    for (int j = 0; j < 8; j++) {
      int rr = (lane < 16) ? j : 8 + j;
      Sc[rr][key] = acc[j] + bp[(size_t)rr * LC];
    }
  }
  __syncthreads();

  // softmax: 2 rows per wave
  for (int rr = wave * 2; rr < wave * 2 + 2; rr++) {
    float m = -3.4e38f;
    for (int k = lane; k < LC; k += 32) m = fmaxf(m, Sc[rr][k]);
    for (int o = 16; o; o >>= 1) m = fmaxf(m, __shfl_xor(m, o, 32));
    float sum = 0.f;
    for (int k = lane; k < 608; k += 32) { float e = __expf(Sc[rr][k] - m); Sc[rr][k] = e; sum += e; }
    for (int o = 16; o; o >>= 1) sum += __shfl_xor(sum, o, 32);
    float inv = 1.0f / sum;
    for (int k = lane; k < 608; k += 32) Pm[rr][k] = to_bf16(Sc[rr][k] * inv);
  }
  __syncthreads();

  // O = P @ V : waves 0..3 each own a 16-wide slice of DH
  if (wave < 4) {
    v8f acc = zero8();
    const int dh = h * DHH + wave * 16 + r;
    const bf16_t* vp = Vt + ((size_t)b * EE + dh) * LT + g * 8;
    for (int ks = 0; ks < 19; ks++) {
      int k = ks * 32;
      ABfrag a, bf;
      a.h[0]  = *(const v8bf*)&Pm[r][k + g * 8];
      a.h[1]  = *(const v8bf*)&Pm[r][k + 16 + g * 8];
      bf.h[0] = *(const v8bf*)(vp + k);
      bf.h[1] = *(const v8bf*)(vp + k + 16);
      acc = wmma_bf16(a.v, bf.v, acc);
    }
    for (int j = 0; j < 8; j++) {
      int rr = (lane < 16) ? j : 8 + j;
      int grow = b * LT + qrow0 + rr;
      Ob[(size_t)grow * EE + dh] = to_bf16(acc[j]);
    }
  }
}

// ---------------- gather the 80 query rows per batch for the LM head ----------------
__global__ __launch_bounds__(256) void gather_q_kernel(const bf16_t* __restrict__ Xb,
                                                       bf16_t* __restrict__ Qlmb) {
  int rr = blockIdx.x;                  // 0..319
  int b = rr / PP, i = rr % PP;
  const bf16_t* src = Xb + ((size_t)(b * LT + LC + i)) * EE;
  bf16_t* dst = Qlmb + (size_t)rr * EE;
  for (int c = threadIdx.x; c < EE; c += blockDim.x) dst[c] = src[c];
}

// ================= host orchestration =================
extern "C" void kernel_launch(void* const* d_in, const int* in_sizes, int n_in,
                              void* d_out, int out_size, void* d_ws, size_t ws_size,
                              hipStream_t stream) {
  (void)in_sizes; (void)n_in; (void)out_size; (void)ws_size;
  const int*   input_ids = (const int*)  d_in[0];
  const float* tok_emb   = (const float*)d_in[1];
  const float* pos_emb   = (const float*)d_in[2];
  const float* emb_g     = (const float*)d_in[3];
  const float* emb_b     = (const float*)d_in[4];
  const float* rel_bias  = (const float*)d_in[5];
  const float* Wq = (const float*)d_in[6];  const float* bq = (const float*)d_in[7];
  const float* Wk = (const float*)d_in[8];  const float* bk = (const float*)d_in[9];
  const float* Wv = (const float*)d_in[10]; const float* bv = (const float*)d_in[11];
  const float* Wo = (const float*)d_in[12]; const float* bo = (const float*)d_in[13];
  const float* ln1_g = (const float*)d_in[14]; const float* ln1_b = (const float*)d_in[15];
  const float* W1 = (const float*)d_in[16]; const float* b1 = (const float*)d_in[17];
  const float* W2 = (const float*)d_in[18]; const float* b2 = (const float*)d_in[19];
  const float* ln2_g = (const float*)d_in[20]; const float* ln2_b = (const float*)d_in[21];
  const float* lm_W = (const float*)d_in[22]; const float* lm_b = (const float*)d_in[23];
  const float* lm_ln_g = (const float*)d_in[24]; const float* lm_ln_b = (const float*)d_in[25];
  const float* lm_bias = (const float*)d_in[26];

  char* ws = (char*)d_ws;
  size_t off = 0;
  auto walloc = [&](size_t bytes) -> void* {
    void* p = ws + off;
    off += (bytes + 255) & ~(size_t)255;
    return p;
  };

  const long nWE  = (long)NL * EE * EE;     // per-stack E x E weights
  const long nWF  = (long)NL * FFNN * EE;   // per-stack FFN weights
  const long nLmW = (long)EE * EE;
  const long nTok = (long)VV * EE;

  float*  X    = (float*) walloc((size_t)MT * EE * 4);
  float*  T    = (float*) walloc((size_t)MT * EE * 4);
  float*  G    = (float*) walloc((size_t)BB * PP * EE * 4);
  float*  biasM= (float*) walloc((size_t)HH * LT * LC * 4);
  bf16_t* Xb   = (bf16_t*)walloc((size_t)MT * EE * 2);
  bf16_t* Qb   = (bf16_t*)walloc((size_t)MT * EE * 2);
  bf16_t* Kb   = (bf16_t*)walloc((size_t)MT * EE * 2);
  bf16_t* Vt   = (bf16_t*)walloc((size_t)BB * EE * LT * 2);
  bf16_t* Obf  = (bf16_t*)walloc((size_t)MT * EE * 2);
  bf16_t* Fb   = (bf16_t*)walloc((size_t)MT * FFNN * 2);
  bf16_t* Qlmb = (bf16_t*)walloc((size_t)BB * PP * EE * 2);
  bf16_t* Hb   = (bf16_t*)walloc((size_t)BB * PP * EE * 2);
  bf16_t* WqB  = (bf16_t*)walloc((size_t)nWE * 2);
  bf16_t* WkB  = (bf16_t*)walloc((size_t)nWE * 2);
  bf16_t* WvB  = (bf16_t*)walloc((size_t)nWE * 2);
  bf16_t* WoB  = (bf16_t*)walloc((size_t)nWE * 2);
  bf16_t* W1B  = (bf16_t*)walloc((size_t)nWF * 2);
  bf16_t* W2B  = (bf16_t*)walloc((size_t)nWF * 2);
  bf16_t* lmWB = (bf16_t*)walloc((size_t)nLmW * 2);
  bf16_t* tokB = (bf16_t*)walloc((size_t)nTok * 2);

  auto cvt = [&](const float* s, bf16_t* d, long n) {
    int blocks = (int)((n + 255) / 256); if (blocks > 4096) blocks = 4096;
    cvt_bf16_kernel<<<blocks, 256, 0, stream>>>(s, d, n);
  };
  cvt(Wq, WqB, nWE); cvt(Wk, WkB, nWE); cvt(Wv, WvB, nWE); cvt(Wo, WoB, nWE);
  cvt(W1, W1B, nWF); cvt(W2, W2B, nWF);
  cvt(lm_W, lmWB, nLmW); cvt(tok_emb, tokB, nTok);

  bias_kernel<<<HH * LT, 256, 0, stream>>>(rel_bias, biasM);
  embed_kernel<<<MT, 256, 0, stream>>>(input_ids, tok_emb, pos_emb, X);
  ln_kernel<<<MT, 256, 0, stream>>>(X, emb_g, emb_b, X, Xb);

  auto gemm = [&](const bf16_t* A, int lda, const bf16_t* W, int ldw, const float* bias,
                  const float* res, float* oF, int ldo, bf16_t* oB, int bmode,
                  int M, int N, int K, float sc, int act) {
    dim3 grid((N + 255) / 256, M / 64);
    gemm_kernel<<<grid, 256, 0, stream>>>(A, lda, W, ldw, bias, res, oF, ldo, oB, bmode,
                                          M, N, K, sc, act);
  };

  for (int l = 0; l < NL; l++) {
    const size_t wE = (size_t)l * EE * EE;
    const size_t wF = (size_t)l * FFNN * EE;
    // Q/K/V projections (Q pre-scaled; V stored transposed for P@V)
    gemm(Xb, EE, WqB + wE, EE, bq + l * EE, nullptr, nullptr, EE, Qb, 1, MT, EE, EE, SCALE_Q, 0);
    gemm(Xb, EE, WkB + wE, EE, bk + l * EE, nullptr, nullptr, EE, Kb, 1, MT, EE, EE, 1.0f, 0);
    gemm(Xb, EE, WvB + wE, EE, bv + l * EE, nullptr, nullptr, EE, Vt, 2, MT, EE, EE, 1.0f, 0);
    // fused attention
    attn_kernel<<<dim3(LT / 16, HH, BB), 256, 0, stream>>>(Qb, Kb, Vt, biasM, Obf);
    // output projection + residual, then LN1
    gemm(Obf, EE, WoB + wE, EE, bo + l * EE, X, T, EE, nullptr, 0, MT, EE, EE, 1.0f, 0);
    ln_kernel<<<MT, 256, 0, stream>>>(T, ln1_g + l * EE, ln1_b + l * EE, X, Xb);
    // FFN
    gemm(Xb, EE, W1B + wF, EE, b1 + l * FFNN, nullptr, nullptr, FFNN, Fb, 1, MT, FFNN, EE, 1.0f, 1);
    gemm(Fb, FFNN, W2B + wF, FFNN, b2 + l * EE, X, T, EE, nullptr, 0, MT, EE, FFNN, 1.0f, 0);
    ln_kernel<<<MT, 256, 0, stream>>>(T, ln2_g + l * EE, ln2_b + l * EE, X, Xb);
  }

  // LM head on the 320 query rows
  gather_q_kernel<<<BB * PP, 256, 0, stream>>>(Xb, Qlmb);
  gemm(Qlmb, EE, lmWB, EE, lm_b, nullptr, G, EE, nullptr, 0, BB * PP, EE, EE, 1.0f, 1);
  ln_kernel<<<BB * PP, 256, 0, stream>>>(G, lm_ln_g, lm_ln_b, nullptr, Hb);
  gemm(Hb, EE, tokB, EE, lm_bias, nullptr, (float*)d_out, VV, nullptr, 0, BB * PP, VV, EE, 1.0f, 0);
}
